// HALT_41120016892315
// MI455X (gfx1250) — compile-verified, matrix operand order checked by reading, not run on hardware
//
#include <hip/hip_runtime.h>
#include <hip/hip_bf16.h>
#include <math.h>

// Problem constants (match reference)
static constexpr int kB   = 256;   // batch
static constexpr int kT   = 512;   // time
static constexpr int kDIN = 25;
static constexpr int kP   = 128;
static constexpr int kH   = 256;   // hidden
static constexpr int kNG  = 768;   // 3*H gates
static constexpr int kNL  = 5;     // layers
static constexpr float kQ = 0.15f;

typedef __bf16 bf16_t;
typedef __attribute__((ext_vector_type(8)))  __bf16 v8bf;
typedef __attribute__((ext_vector_type(16))) __bf16 v16bf;
typedef __attribute__((ext_vector_type(8)))  float  v8f;

// ---------------------------------------------------------------------------
// Load a 16x32 bf16 WMMA fragment (A layout; B layout is the mirror with the
// lane indexing the N column of a row-major [N,K] weight).
// lane&15 selects the row; kh = 8*(lane>>4). Elements 0..7 = K[kbase+kh..+7],
// elements 8..15 = K[kbase+16+kh..+7]. Two contiguous 16-byte loads.
// ---------------------------------------------------------------------------
__device__ __forceinline__ v16bf load_frag(const bf16_t* __restrict__ row,
                                           int kbase, int kh) {
  v8bf lo = *reinterpret_cast<const v8bf*>(row + kbase + kh);
  v8bf hi = *reinterpret_cast<const v8bf*>(row + kbase + 16 + kh);
  return __builtin_shufflevector(lo, hi, 0, 1, 2, 3, 4, 5, 6, 7,
                                         8, 9, 10, 11, 12, 13, 14, 15);
}

__device__ __forceinline__ float sigf(float x) {
  return 1.f / (1.f + expf(-x));
}

// ---------------------------------------------------------------------------
// fp32 -> bf16 weight conversion
// ---------------------------------------------------------------------------
__global__ void cvt_bf16_kernel(const float* __restrict__ src,
                                bf16_t* __restrict__ dst, int n) {
  int i = blockIdx.x * blockDim.x + threadIdx.x;
  if (i < n) dst[i] = (bf16_t)src[i];
}

// ---------------------------------------------------------------------------
// Zero the GRU hidden state t=0 input buffers (fp32 + bf16), both directions
// ---------------------------------------------------------------------------
__global__ void hzero_kernel(float* __restrict__ h, bf16_t* __restrict__ hb) {
  int i = blockIdx.x * blockDim.x + threadIdx.x;
  if (i < 2 * kB * kH) { h[i] = 0.f; hb[i] = (bf16_t)0.f; }
}

// ---------------------------------------------------------------------------
// Input projection: LayerNorm(25) -> Linear(25,128) -> exact GELU -> Linear(128,128)
// One block per token, 128 threads. Output bf16 [B*T, 128].
// ---------------------------------------------------------------------------
__global__ void __launch_bounds__(128)
input_proj_kernel(const float* __restrict__ x, const float* __restrict__ ln_g,
                  const float* __restrict__ ln_b, const float* __restrict__ w1,
                  const float* __restrict__ b1, const float* __restrict__ w2,
                  const float* __restrict__ b2, bf16_t* __restrict__ out) {
  __shared__ float xs[kDIN];
  __shared__ float h1[kP];
  const int tok = blockIdx.x;
  const int tid = threadIdx.x;
  const float* xr = x + (size_t)tok * kDIN;
  if (tid < kDIN) xs[tid] = xr[tid];
  __syncthreads();

  float mu = 0.f;
#pragma unroll
  for (int d = 0; d < kDIN; ++d) mu += xs[d];
  mu *= (1.0f / kDIN);
  float var = 0.f;
#pragma unroll
  for (int d = 0; d < kDIN; ++d) { float t = xs[d] - mu; var += t * t; }
  var *= (1.0f / kDIN);
  float inv = rsqrtf(var + 1e-5f);

  float acc = b1[tid];
#pragma unroll
  for (int d = 0; d < kDIN; ++d) {
    float xn = (xs[d] - mu) * inv * ln_g[d] + ln_b[d];
    acc += xn * w1[d * kP + tid];
  }
  // exact GELU
  float g = 0.5f * acc * (1.f + erff(acc * 0.70710678118654752f));
  h1[tid] = g;
  __syncthreads();

  float acc2 = b2[tid];
#pragma unroll 8
  for (int p = 0; p < kP; ++p) acc2 += h1[p] * w2[p * kP + tid];
  out[(size_t)tok * kP + tid] = (bf16_t)acc2;
}

// ---------------------------------------------------------------------------
// Fully-fused GRU time step, both directions, WMMA bf16 -> f32.
// Each wave owns one 16x16 (batch x hidden) tile of ONE direction and computes
// all three gates for it:
//   gx_{r,z,n} = x_t @ Wih_{r,z,n}^T     (K = din; bwd walks reversed prefix)
//   gh_{r,z,n} = h_in @ Whh_{r,z,n}^T    (K = H)
// The A fragment (x row / h row) is reused across the 3 gate WMMAs.
// Epilogue applies sigmoid/tanh gate math, masking, state hold, and scatters
// y with per-sample reversal. h is double-buffered (h_in read-only this step,
// h_out written) so there is no intra-step RAW hazard across waves.
// Mapping: 2 dirs * (16 x 16) tiles = 512 waves = 64 blocks of 8 waves.
// ---------------------------------------------------------------------------
__global__ void __launch_bounds__(256)
gru_step_fused_kernel(const bf16_t* __restrict__ xin, int ld_in, int din,
                      const bf16_t* __restrict__ hb_in,
                      bf16_t* __restrict__ hb_out,
                      const float* __restrict__ h32_in,
                      float* __restrict__ h32_out,
                      const bf16_t* __restrict__ Wih,
                      const bf16_t* __restrict__ Whh,
                      const float* __restrict__ bih,
                      const float* __restrict__ bhh,
                      const int* __restrict__ lengths, int t,
                      bf16_t* __restrict__ yout) {
  const int wv   = threadIdx.x >> 5;
  const int lane = threadIdx.x & 31;
  const int gw   = blockIdx.x * 8 + wv;
  const int dir  = gw >> 8;            // 0 = fwd, 1 = bwd
  const int tile = gw & 255;
  const int tm   = tile >> 4;          // batch tile
  const int tn   = tile & 15;          // hidden tile

  const int mrow = lane & 15;          // A row (M) / B column (N) for this lane
  const int kh   = (lane >> 4) << 3;   // K half offset: 0 or 8

  // A-side row (x and h): per-sample time index for the backward direction
  const int ba   = tm * 16 + mrow;
  const int lena = lengths[ba];
  int tt = dir ? (lena - 1 - t) : t;
  if (tt < 0) tt = 0;
  const bf16_t* xrow = xin + ((size_t)ba * kT + tt) * ld_in;
  const bf16_t* hrow = hb_in + ((size_t)dir * kB + ba) * kH;

  // B-side: weight rows for the three gates sharing hidden index jj
  const int jj = tn * 16 + mrow;
  const bf16_t* wr = Wih + ((size_t)dir * kNG + jj) * din;            // r
  const bf16_t* wz = wr + (size_t)kH * din;                           // z
  const bf16_t* wn = wr + (size_t)2 * kH * din;                       // n
  const bf16_t* ur = Whh + ((size_t)dir * kNG + jj) * kH;             // r
  const bf16_t* uz = ur + (size_t)kH * kH;                            // z
  const bf16_t* un = ur + (size_t)2 * kH * kH;                        // n

  v8f axr = {}, axz = {}, axn = {};
  for (int k = 0; k < din; k += 32) {
    v16bf a = load_frag(xrow, k, kh);
    axr = __builtin_amdgcn_wmma_f32_16x16x32_bf16(
        false, a, false, load_frag(wr, k, kh), (short)0, axr, false, false);
    axz = __builtin_amdgcn_wmma_f32_16x16x32_bf16(
        false, a, false, load_frag(wz, k, kh), (short)0, axz, false, false);
    axn = __builtin_amdgcn_wmma_f32_16x16x32_bf16(
        false, a, false, load_frag(wn, k, kh), (short)0, axn, false, false);
  }
  v8f ahr = {}, ahz = {}, ahn = {};
#pragma unroll
  for (int k = 0; k < kH; k += 32) {
    v16bf a = load_frag(hrow, k, kh);
    ahr = __builtin_amdgcn_wmma_f32_16x16x32_bf16(
        false, a, false, load_frag(ur, k, kh), (short)0, ahr, false, false);
    ahz = __builtin_amdgcn_wmma_f32_16x16x32_bf16(
        false, a, false, load_frag(uz, k, kh), (short)0, ahz, false, false);
    ahn = __builtin_amdgcn_wmma_f32_16x16x32_bf16(
        false, a, false, load_frag(un, k, kh), (short)0, ahn, false, false);
  }

  // Biases: per gate column jj (same across the 8 rows this lane holds)
  const float bir = bih[dir * kNG + jj];
  const float biz = bih[dir * kNG + kH + jj];
  const float bin = bih[dir * kNG + 2 * kH + jj];
  const float bhr = bhh[dir * kNG + jj];
  const float bhz = bhh[dir * kNG + kH + jj];
  const float bhn = bhh[dir * kNG + 2 * kH + jj];

  // D layout: lane holds N = jj, VGPR r holds row M = tm*16 + r + 8*(lane>>4)
  const int mh = (lane >> 4) * 8;
#pragma unroll
  for (int r = 0; r < 8; ++r) {
    const int brow = tm * 16 + mh + r;
    const int len  = lengths[brow];
    const size_t hi = ((size_t)dir * kB + brow) * kH + jj;
    if (t < len) {
      float rg  = sigf(axr[r] + ahr[r] + bir + bhr);
      float zg  = sigf(axz[r] + ahz[r] + biz + bhz);
      float ghn = ahn[r] + bhn;
      float ng  = tanhf(axn[r] + bin + rg * ghn);
      float hp  = h32_in[hi];
      float hv  = (1.f - zg) * ng + zg * hp;
      h32_out[hi] = hv;
      hb_out[hi]  = (bf16_t)hv;
      int to = dir ? (len - 1 - t) : t;
      yout[((size_t)brow * kT + to) * (2 * kH) + dir * kH + jj] = (bf16_t)hv;
    } else {
      // hold state past sequence end
      h32_out[hi] = h32_in[hi];
      hb_out[hi]  = hb_in[hi];
      if (dir == 0) {
        // padded output rows are zero (cover both direction halves)
        yout[((size_t)brow * kT + t) * (2 * kH) + jj]      = (bf16_t)0.f;
        yout[((size_t)brow * kT + t) * (2 * kH) + kH + jj] = (bf16_t)0.f;
      }
    }
  }
}

// ---------------------------------------------------------------------------
// TopQ pooling + classifier head. One block (256 threads) per sample.
// scores = L2 norm of H[b,t,:], mask invalid; iterative argmax k times.
// ---------------------------------------------------------------------------
__global__ void __launch_bounds__(256)
topq_pool_kernel(const bf16_t* __restrict__ H, const int* __restrict__ lengths,
                 const float* __restrict__ Wc, const float* __restrict__ bc,
                 float* __restrict__ out) {
  __shared__ float sc[kT];
  __shared__ float red[256];
  __shared__ int   redi[256];
  const int b = blockIdx.x;
  const int tid = threadIdx.x;
  const int len = lengths[b];

  for (int t = tid; t < kT; t += 256) {
    float s;
    if (t < len) {
      const bf16_t* hr = H + ((size_t)b * kT + t) * (2 * kH);
      float acc = 0.f;
      for (int d = 0; d < 2 * kH; ++d) { float v = (float)hr[d]; acc += v * v; }
      s = sqrtf(acc);
    } else {
      s = -1e9f;
    }
    sc[t] = s;
  }
  __syncthreads();

  int k = (int)ceilf((float)len * kQ);
  if (k < 1) k = 1;

  float a0 = 0.f, a1 = 0.f;
  for (int it = 0; it < k; ++it) {
    float best = sc[tid];
    int   bi   = tid;
    float s2   = sc[tid + 256];
    if (s2 > best) { best = s2; bi = tid + 256; }
    red[tid] = best; redi[tid] = bi;
    __syncthreads();
    for (int off = 128; off > 0; off >>= 1) {
      if (tid < off && red[tid + off] > red[tid]) {
        red[tid] = red[tid + off]; redi[tid] = redi[tid + off];
      }
      __syncthreads();
    }
    int tmax = redi[0];
    __syncthreads();
    if (tid == 0) sc[tmax] = -2e9f;
    const bf16_t* hr = H + ((size_t)b * kT + tmax) * (2 * kH);
    a0 += (float)hr[tid];
    a1 += (float)hr[tid + 256];
    __syncthreads();
  }

  float invk = 1.f / (float)k;
  float part = a0 * invk * Wc[tid] + a1 * invk * Wc[tid + 256];
  red[tid] = part;
  __syncthreads();
  for (int off = 128; off > 0; off >>= 1) {
    if (tid < off) red[tid] += red[tid + off];
    __syncthreads();
  }
  if (tid == 0) out[b] = red[0] + bc[0];
}

// ---------------------------------------------------------------------------
// Host-side orchestration
// ---------------------------------------------------------------------------
extern "C" void kernel_launch(void* const* d_in, const int* in_sizes, int n_in,
                              void* d_out, int out_size, void* d_ws,
                              size_t ws_size, hipStream_t stream) {
  (void)in_sizes; (void)n_in; (void)out_size; (void)ws_size;

  const float* x       = (const float*)d_in[0];
  const int*   lengths = (const int*)d_in[1];
  const float* ln_g    = (const float*)d_in[2];
  const float* ln_b    = (const float*)d_in[3];
  const float* w1      = (const float*)d_in[4];
  const float* b1      = (const float*)d_in[5];
  const float* w2      = (const float*)d_in[6];
  const float* b2      = (const float*)d_in[7];
  const float* Wih0    = (const float*)d_in[8];
  const float* Whh0    = (const float*)d_in[9];
  const float* bih0    = (const float*)d_in[10];
  const float* bhh0    = (const float*)d_in[11];
  const float* Wih     = (const float*)d_in[12];
  const float* Whh     = (const float*)d_in[13];
  const float* bih     = (const float*)d_in[14];
  const float* bhh     = (const float*)d_in[15];
  const float* Wc      = (const float*)d_in[16];
  const float* bc      = (const float*)d_in[17];
  float* out = (float*)d_out;

  char* ws = (char*)d_ws;
  size_t off = 0;
  auto take = [&](size_t bytes) -> char* {
    char* p = ws + off;
    off = (off + bytes + 255) & ~(size_t)255;
    return p;
  };

  bf16_t* bufA  = (bf16_t*)take((size_t)kB * kT * (2 * kH) * sizeof(bf16_t));
  bf16_t* bufB  = (bf16_t*)take((size_t)kB * kT * (2 * kH) * sizeof(bf16_t));
  bf16_t* wih0b = (bf16_t*)take((size_t)2 * kNG * kP * sizeof(bf16_t));
  bf16_t* whh0b = (bf16_t*)take((size_t)2 * kNG * kH * sizeof(bf16_t));
  bf16_t* wihb  = (bf16_t*)take((size_t)4 * 2 * kNG * (2 * kH) * sizeof(bf16_t));
  bf16_t* whhb  = (bf16_t*)take((size_t)4 * 2 * kNG * kH * sizeof(bf16_t));
  float*  h32a  = (float*)take((size_t)2 * kB * kH * sizeof(float));
  float*  h32b  = (float*)take((size_t)2 * kB * kH * sizeof(float));
  bf16_t* hbfa  = (bf16_t*)take((size_t)2 * kB * kH * sizeof(bf16_t));
  bf16_t* hbfb  = (bf16_t*)take((size_t)2 * kB * kH * sizeof(bf16_t));

  // Convert weights to bf16 once per call
  {
    int n0 = 2 * kNG * kP;
    int n1 = 2 * kNG * kH;
    int n2 = 4 * 2 * kNG * (2 * kH);
    int n3 = 4 * 2 * kNG * kH;
    cvt_bf16_kernel<<<(n0 + 255) / 256, 256, 0, stream>>>(Wih0, wih0b, n0);
    cvt_bf16_kernel<<<(n1 + 255) / 256, 256, 0, stream>>>(Whh0, whh0b, n1);
    cvt_bf16_kernel<<<(n2 + 255) / 256, 256, 0, stream>>>(Wih, wihb, n2);
    cvt_bf16_kernel<<<(n3 + 255) / 256, 256, 0, stream>>>(Whh, whhb, n3);
  }

  // Input projection -> bufA as [B*T, 128] bf16
  input_proj_kernel<<<kB * kT, 128, 0, stream>>>(x, ln_g, ln_b, w1, b1, w2, b2,
                                                 bufA);

  const bf16_t* in = bufA;
  int ld_in = kP;
  bf16_t* outb = bufB;
  for (int l = 0; l < kNL; ++l) {
    // zero the t=0 hidden-state input buffers
    hzero_kernel<<<(2 * kB * kH + 255) / 256, 256, 0, stream>>>(h32a, hbfa);
    int din = (l == 0) ? kP : (2 * kH);
    const bf16_t* wi = (l == 0) ? wih0b : wihb + (size_t)(l - 1) * 2 * kNG * (2 * kH);
    const bf16_t* wh = (l == 0) ? whh0b : whhb + (size_t)(l - 1) * 2 * kNG * kH;
    const float*  bi = (l == 0) ? bih0 : bih + (size_t)(l - 1) * 2 * kNG;
    const float*  bh = (l == 0) ? bhh0 : bhh + (size_t)(l - 1) * 2 * kNG;
    for (int t = 0; t < kT; ++t) {
      const bool odd = (t & 1) != 0;
      const bf16_t* hbi = odd ? hbfb : hbfa;
      bf16_t*       hbo = odd ? hbfa : hbfb;
      const float*  h3i = odd ? h32b : h32a;
      float*        h3o = odd ? h32a : h32b;
      gru_step_fused_kernel<<<64, 256, 0, stream>>>(
          in, ld_in, din, hbi, hbo, h3i, h3o, wi, wh, bi, bh, lengths, t, outb);
    }
    in = outb;
    ld_in = 2 * kH;
    outb = (outb == bufB) ? bufA : bufB;
  }

  // Final H is in `in` (bufB after 5 layers)
  topq_pool_kernel<<<kB, 256, 0, stream>>>(in, lengths, Wc, bc, out);
}
